// nn_Block_29317446762759
// MI455X (gfx1250) — matrix operand order repaired and checked by reading, not compiled, not run on hardware
//
#include <hip/hip_runtime.h>
#include <hip/hip_bf16.h>

typedef __attribute__((ext_vector_type(16))) _Float16 v16h;
typedef __attribute__((ext_vector_type(8)))  _Float16 v8h;
typedef __attribute__((ext_vector_type(8)))  float    v8f;

#define NTOK 2048
#define MD_   64
#define NHN   8
#define HEADS 4
#define DH    16

// Build a 16-bit A-fragment (contiguous K layout): per lane, elements 0..15 are
// K = bk..bk+15 (bk = 16*(lane/16)) -> two
// contiguous 16B LDS loads.
__device__ __forceinline__ v16h frag_from2(const _Float16* p0, const _Float16* p1) {
  v8h lo = *(const v8h*)p0;
  v8h hi = *(const v8h*)p1;
  return __builtin_shufflevector(lo, hi, 0, 1, 2, 3, 4, 5, 6, 7,
                                 8, 9, 10, 11, 12, 13, 14, 15);
}

// ---------------------------------------------------------------------------
// 1) top-k (8 nearest by sqrt(lon^2+lat^2), ties -> lower index)
// ---------------------------------------------------------------------------
__global__ __launch_bounds__(256) void topk_kernel(const float* __restrict__ lon,
                                                   const float* __restrict__ lat,
                                                   int* __restrict__ idx, int N) {
  __shared__ float bv[256];
  __shared__ int   bi[256];
  __shared__ int   sel[NHN];
  const int n = blockIdx.x, t = threadIdx.x;
  const float* ln = lon + (size_t)n * N;
  const float* lt = lat + (size_t)n * N;
  for (int it = 0; it < NHN; ++it) {
    float best = __builtin_inff();
    int   bidx = N;
    for (int j = t; j < N; j += 256) {
      bool taken = false;
      for (int s = 0; s < it; ++s) taken |= (sel[s] == j);
      if (taken) continue;
      float d = ln[j] * ln[j] + lt[j] * lt[j];  // monotone in sqrt
      if (d < best || (d == best && j < bidx)) { best = d; bidx = j; }
    }
    bv[t] = best; bi[t] = bidx;
    __syncthreads();
    for (int s = 128; s > 0; s >>= 1) {
      if (t < s) {
        if (bv[t + s] < bv[t] || (bv[t + s] == bv[t] && bi[t + s] < bi[t])) {
          bv[t] = bv[t + s]; bi[t] = bi[t + s];
        }
      }
      __syncthreads();
    }
    if (t == 0) { sel[it] = bi[0]; idx[(size_t)n * NHN + it] = bi[0]; }
    __syncthreads();
  }
}

// ---------------------------------------------------------------------------
// 2) gather + conv(4->64) + LN over neighbor axis + leaky-relu + max-pool
// ---------------------------------------------------------------------------
__global__ __launch_bounds__(64) void gather_kernel(const float* __restrict__ x,
                                                    const float* __restrict__ cab_lon,
                                                    const int* __restrict__ idx,
                                                    const float* __restrict__ conv_w,
                                                    const float* __restrict__ g,
                                                    const float* __restrict__ bta,
                                                    float* __restrict__ x2, int N) {
  __shared__ float xg[NHN];
  __shared__ float dl[NHN];
  const int bn = blockIdx.x;
  const int b = bn / N, n = bn % N, t = threadIdx.x;
  if (t < NHN) {
    int i = idx[(size_t)n * NHN + t];
    xg[t] = x[(size_t)b * N + i];  // E == 1
    dl[t] = cab_lon[i];
  }
  __syncthreads();
  const float x0 = xg[0], d0 = dl[0];
  float hv[NHN], mean = 0.f;
#pragma unroll
  for (int k = 0; k < NHN; ++k) {
    float f0 = xg[k] - x0, f1 = dl[k] - d0;  // c1 == c2 == d_lon in the ref
    float h = f0 * conv_w[t] + f1 * conv_w[64 + t] + f1 * conv_w[128 + t] + x0 * conv_w[192 + t];
    hv[k] = h; mean += h;
  }
  mean *= 0.125f;
  float var = 0.f;
#pragma unroll
  for (int k = 0; k < NHN; ++k) { float dd = hv[k] - mean; var += dd * dd; }
  var *= 0.125f;
  const float inv = rsqrtf(var + 1e-5f);
  float mx = -__builtin_inff();
#pragma unroll
  for (int k = 0; k < NHN; ++k) {
    float y = (hv[k] - mean) * inv * g[k] + bta[k];
    y = (y > 0.f) ? y : 0.2f * y;
    mx = fmaxf(mx, y);
  }
  x2[(size_t)bn * MD_ + t] = mx;
}

// ---------------------------------------------------------------------------
// 3) RPE MLP -> rel_emb (HEADS, N, N), 64 MB, L2-resident
// ---------------------------------------------------------------------------
__global__ __launch_bounds__(256) void rpe_kernel(const float* __restrict__ lon,
                                                  const float* __restrict__ lat,
                                                  const float* __restrict__ w1,
                                                  const float* __restrict__ b1,
                                                  const float* __restrict__ w2,
                                                  const float* __restrict__ b2,
                                                  float* __restrict__ rel, int N) {
  __shared__ float sw1[32], sb1[16], sw2[64], sb2[4];
  const int t = threadIdx.x;
  if (t < 32) sw1[t] = w1[t];
  if (t < 16) sb1[t] = b1[t];
  if (t < 64) sw2[t] = w2[t];
  if (t < 4)  sb2[t] = b2[t];
  __syncthreads();
  const size_t NN = (size_t)N * N;
  const size_t id = (size_t)blockIdx.x * 256 + t;
  if (id >= NN) return;
  const float lo = lon[id], la = lat[id];
  float o0 = sb2[0], o1 = sb2[1], o2 = sb2[2], o3 = sb2[3];
#pragma unroll
  for (int tt = 0; tt < 16; ++tt) {
    float hd = fmaxf(lo * sw1[tt] + la * sw1[16 + tt] + sb1[tt], 0.f);
    o0 += hd * sw2[tt * 4 + 0]; o1 += hd * sw2[tt * 4 + 1];
    o2 += hd * sw2[tt * 4 + 2]; o3 += hd * sw2[tt * 4 + 3];
  }
  rel[0 * NN + id] = o0; rel[1 * NN + id] = o1;
  rel[2 * NN + id] = o2; rel[3 * NN + id] = o3;
}

// ---------------------------------------------------------------------------
// 4) WMMA GEMM: Y[M,64] = act(X[M,64] @ W[64,64] + bias)
//    128-row block tile, 8 waves; X staged row-major f16, W staged TRANSPOSED
//    (wT[n][k]) so B-fragments are contiguous v16h LDS loads.
// ---------------------------------------------------------------------------
__global__ __launch_bounds__(256) void gemm64_kernel(const float* __restrict__ X,
                                                     const float* __restrict__ W,
                                                     const float* __restrict__ bias,
                                                     float* __restrict__ Y,
                                                     int M, int act) {
  __shared__ __attribute__((aligned(64))) _Float16 xs[128][64];
  __shared__ __attribute__((aligned(64))) _Float16 wT[64][64];  // wT[n][k]
  const int tid = threadIdx.x;
  const int rowbase = blockIdx.x * 128;
  // X tile: 2048 float4 loads spread over 256 threads
  const float4* X4 = (const float4*)(X + (size_t)rowbase * 64);
  for (int t = tid; t < 128 * 16; t += 256) {
    float4 v = X4[t];
    int r = t >> 4, c = (t & 15) * 4;
    xs[r][c + 0] = (_Float16)v.x; xs[r][c + 1] = (_Float16)v.y;
    xs[r][c + 2] = (_Float16)v.z; xs[r][c + 3] = (_Float16)v.w;
  }
  for (int t = tid; t < 64 * 64; t += 256) {
    int k = t >> 6, n = t & 63;
    wT[n][k] = (_Float16)W[t];
  }
  __syncthreads();

  const int wave = tid >> 5, lane = tid & 31;
  const int r0 = wave * 16;
  const int arow = r0 + (lane & 15);
  const int half8 = (lane >> 4) << 3;      // C row offset
  const int bcol = lane & 15;
  const int bk = (lane >> 4) * 16;         // B-frag K offset

  v16h a0 = frag_from2(&xs[arow][bk], &xs[arow][bk + 8]);
  v16h a1 = frag_from2(&xs[arow][bk + 32], &xs[arow][bk + 40]);

#pragma unroll
  for (int nt = 0; nt < 4; ++nt) {
    v16h b0 = *(const v16h*)&wT[nt * 16 + bcol][bk];
    v16h b1 = *(const v16h*)&wT[nt * 16 + bcol][bk + 32];
    v8f c = {0.f, 0.f, 0.f, 0.f, 0.f, 0.f, 0.f, 0.f};
    c = __builtin_amdgcn_wmma_f32_16x16x32_f16(false, a0, false, b0, (short)0, c, false, false);
    c = __builtin_amdgcn_wmma_f32_16x16x32_f16(false, a1, false, b1, (short)0, c, false, false);
#pragma unroll
    for (int r = 0; r < 8; ++r) {
      int row = r + half8;
      int col = nt * 16 + bcol;
      float v = c[r] + bias[col];
      if (act == 1) v = fmaxf(v, 0.f);
      Y[(size_t)(rowbase + r0 + row) * 64 + col] = v;
    }
  }
}

// ---------------------------------------------------------------------------
// 5) Flash attention per (b, h): 64 query rows/block, 64-key tiles.
//    S = (scale*Qn) Kn^T via WMMA, + rel_emb, online softmax, O += P V via WMMA.
//    V staged transposed (vsT[d][key]) for contiguous B-fragment loads.
// ---------------------------------------------------------------------------
__global__ __launch_bounds__(128) void attn_kernel(const float* __restrict__ Q,
                                                   const float* __restrict__ K,
                                                   const float* __restrict__ V,
                                                   const float* __restrict__ rel,
                                                   const float* __restrict__ logit_scale,
                                                   float* __restrict__ O, int N) {
  __shared__ __attribute__((aligned(64))) _Float16 qs[64][32];
  __shared__ __attribute__((aligned(64))) _Float16 ks[64][32];   // ks[key][d]
  __shared__ __attribute__((aligned(64))) _Float16 vsT[16][64];  // vsT[d][key]
  __shared__ __attribute__((aligned(64))) _Float16 ps[64][64];
  __shared__ float stile[64][64];
  __shared__ float mrow[64], lrow[64], arow_s[64];

  const int tid = threadIdx.x;
  const int qt = blockIdx.x, h = blockIdx.y, b = blockIdx.z;
  const int qbase = qt * 64;
  const float scale = __expf(fminf(logit_scale[h], 4.6051701860f));  // ln(100)

  if (tid < 64) {
    int gq = qbase + tid;
    float qv[DH]; float s2 = 0.f;
#pragma unroll
    for (int d = 0; d < DH; ++d) {
      qv[d] = Q[((size_t)b * N + gq) * MD_ + h * DH + d];
      s2 += qv[d] * qv[d];
    }
    float inv = scale / (sqrtf(s2) + 1e-6f);   // fold logit scale into Qn
#pragma unroll
    for (int d = 0; d < DH; ++d)  qs[tid][d] = (_Float16)(qv[d] * inv);
#pragma unroll
    for (int d = DH; d < 32; ++d) qs[tid][d] = (_Float16)0.f;
    mrow[tid] = -__builtin_inff();
    lrow[tid] = 0.f;
  }
  __syncthreads();

  const int wave = tid >> 5, lane = tid & 31;
  const int half8 = (lane >> 4) << 3;
  const int bcol = lane & 15;
  const int bk = (lane >> 4) * 16;
  const int arow = wave * 16 + (lane & 15);

  const v16h aq = frag_from2(&qs[arow][bk], &qs[arow][bk + 8]);

  v8f oacc = {0.f, 0.f, 0.f, 0.f, 0.f, 0.f, 0.f, 0.f};

  const int NT = N / 64;
  for (int kt = 0; kt < NT; ++kt) {
    if (tid < 64) {
      int gk = kt * 64 + tid;
      const float* Kp = &K[((size_t)b * N + gk) * MD_ + h * DH];
      const float* Vp = &V[((size_t)b * N + gk) * MD_ + h * DH];
      if (kt + 1 < NT) {  // gfx1250 global_prefetch of next tile
        __builtin_prefetch(Kp + (size_t)64 * MD_, 0, 0);
        __builtin_prefetch(Vp + (size_t)64 * MD_, 0, 0);
      }
      float kv[DH]; float s2 = 0.f;
#pragma unroll
      for (int d = 0; d < DH; ++d) { kv[d] = Kp[d]; s2 += kv[d] * kv[d]; }
      float inv = 1.f / (sqrtf(s2) + 1e-6f);
#pragma unroll
      for (int d = 0; d < DH; ++d)  ks[tid][d] = (_Float16)(kv[d] * inv);
#pragma unroll
      for (int d = DH; d < 32; ++d) ks[tid][d] = (_Float16)0.f;
#pragma unroll
      for (int d = 0; d < DH; ++d)  vsT[d][tid] = (_Float16)Vp[d];
    }
    __syncthreads();

    // scores: 4 tiles of 16 keys; B = Kn^T == contiguous rows of ks[key][d]
#pragma unroll
    for (int nt = 0; nt < 4; ++nt) {
      v16h bf = *(const v16h*)&ks[nt * 16 + bcol][bk];
      v8f c = {0.f, 0.f, 0.f, 0.f, 0.f, 0.f, 0.f, 0.f};
      c = __builtin_amdgcn_wmma_f32_16x16x32_f16(false, aq, false, bf, (short)0, c, false, false);
#pragma unroll
      for (int r = 0; r < 8; ++r) {
        int rowt = wave * 16 + r + half8;
        int colk = nt * 16 + bcol;
        size_t gq = (size_t)qbase + rowt;
        size_t gk = (size_t)kt * 64 + colk;
        stile[rowt][colk] = c[r] + rel[((size_t)h * N + gq) * N + gk];
      }
    }
    __syncthreads();

    if (tid < 64) {  // online softmax row update
      float mo = mrow[tid], mx = mo;
      for (int j = 0; j < 64; ++j) mx = fmaxf(mx, stile[tid][j]);
      float al = __expf(mo - mx);
      float sum = 0.f;
      for (int j = 0; j < 64; ++j) {
        float p = __expf(stile[tid][j] - mx);
        ps[tid][j] = (_Float16)p;
        sum += p;
      }
      lrow[tid] = lrow[tid] * al + sum;
      mrow[tid] = mx;
      arow_s[tid] = al;
    }
    __syncthreads();

    // O = O*alpha + P V
#pragma unroll
    for (int r = 0; r < 8; ++r) oacc[r] *= arow_s[wave * 16 + r + half8];
#pragma unroll
    for (int kc = 0; kc < 2; ++kc) {
      v16h ap = frag_from2(&ps[arow][kc * 32 + bk], &ps[arow][kc * 32 + bk + 8]);
      v16h bv = *(const v16h*)&vsT[bcol][kc * 32 + bk];
      oacc = __builtin_amdgcn_wmma_f32_16x16x32_f16(false, ap, false, bv, (short)0, oacc, false, false);
    }
    __syncthreads();
  }

#pragma unroll
  for (int r = 0; r < 8; ++r) {
    int rowt = wave * 16 + r + half8;
    float v = oacc[r] / lrow[rowt];
    O[((size_t)b * N + qbase + rowt) * MD_ + h * DH + (lane & 15)] = v;
  }
}

// ---------------------------------------------------------------------------
// 6) residual + LayerNorm over 64 features
// ---------------------------------------------------------------------------
__global__ __launch_bounds__(64) void ln_kernel(const float* __restrict__ X,
                                                const float* __restrict__ R,
                                                const float* __restrict__ g,
                                                const float* __restrict__ bta,
                                                float* __restrict__ Y) {
  __shared__ float s1[64], s2[64];
  const int row = blockIdx.x, t = threadIdx.x;
  float v = X[(size_t)row * 64 + t];
  if (R) v += R[(size_t)row * 64 + t];
  s1[t] = v; s2[t] = v * v;
  __syncthreads();
  for (int s = 32; s > 0; s >>= 1) {
    if (t < s) { s1[t] += s1[t + s]; s2[t] += s2[t + s]; }
    __syncthreads();
  }
  float mean = s1[0] * 0.015625f;
  float var = s2[0] * 0.015625f - mean * mean;
  float xn = (v - mean) * rsqrtf(var + 1e-5f);
  Y[(size_t)row * 64 + t] = xn * g[t] + bta[t];
}

// ---------------------------------------------------------------------------
// 7) final 64->1 projection with ReLU (wave reduction)
// ---------------------------------------------------------------------------
__global__ __launch_bounds__(256) void gemv_relu_kernel(const float* __restrict__ H,
                                                        const float* __restrict__ w,
                                                        const float* __restrict__ b,
                                                        float* __restrict__ out) {
  const int wave = threadIdx.x >> 5, lane = threadIdx.x & 31;
  const int row = blockIdx.x * 8 + wave;
  const float* hr = H + (size_t)row * 64;
  float s = hr[lane] * w[lane] + hr[lane + 32] * w[lane + 32];
#pragma unroll
  for (int o = 16; o > 0; o >>= 1) s += __shfl_down(s, o, 32);
  if (lane == 0) out[row] = fmaxf(s + b[0], 0.f);
}

// ---------------------------------------------------------------------------
// host wiring
// ---------------------------------------------------------------------------
struct Blk {
  const float *b1, *b2, *kb, *kw, *ln1_b, *ln1_g, *ln2_b, *ln2_g, *ls,
              *ob, *ow, *qb, *qw, *vb, *vw, *w1, *w2;
};

extern "C" void kernel_launch(void* const* d_in, const int* in_sizes, int n_in,
                              void* d_out, int out_size, void* d_ws, size_t ws_size,
                              hipStream_t stream) {
  const int N = NTOK, B = 4;
  const int M = B * N;                       // 8192 rows
  const size_t SZ = (size_t)M * MD_;         // 524288 floats per activation
  const size_t NN = (size_t)N * N;

  const float* x       = (const float*)d_in[0];
  const float* crn_lon = (const float*)d_in[1];
  const float* crn_lat = (const float*)d_in[2];
  const float* cab_lon = (const float*)d_in[3];
  // d_in[4] = cab_lat (unused by the reference's feature construction)
  const float* crg_lon = (const float*)d_in[5];
  const float* crg_lat = (const float*)d_in[6];

  // params flattened in JAX pytree (sorted-key) order:
  // 7..23 = blocks[0]{b1,b2,kb,kw,ln1_b,ln1_g,ln2_b,ln2_g,logit_scale,ob,ow,qb,qw,vb,vw,w1,w2}
  // 24..40 = blocks[1] same, 41..47 = conv_w, ln_in_b, ln_in_g, rpe_b1, rpe_b2, rpe_w1, rpe_w2
  auto IN = [&](int i) -> const float* { return (const float*)d_in[i]; };
  auto mkblk = [&](int base) {
    Blk k;
    k.b1 = IN(base + 0);  k.b2 = IN(base + 1);  k.kb = IN(base + 2);
    k.kw = IN(base + 3);  k.ln1_b = IN(base + 4); k.ln1_g = IN(base + 5);
    k.ln2_b = IN(base + 6); k.ln2_g = IN(base + 7); k.ls = IN(base + 8);
    k.ob = IN(base + 9);  k.ow = IN(base + 10); k.qb = IN(base + 11);
    k.qw = IN(base + 12); k.vb = IN(base + 13); k.vw = IN(base + 14);
    k.w1 = IN(base + 15); k.w2 = IN(base + 16);
    return k;
  };
  Blk blk0 = mkblk(7), blk1 = mkblk(24);
  const float* conv_w  = IN(41);
  const float* ln_in_b = IN(42);
  const float* ln_in_g = IN(43);
  const float* rpe_b1  = IN(44);
  const float* rpe_b2  = IN(45);
  const float* rpe_w1  = IN(46);
  const float* rpe_w2  = IN(47);

  // workspace layout (~84 MB): idx ints, then float buffers; rel stays in L2
  int*   idxp = (int*)d_ws;
  float* f    = (float*)d_ws + 16384;
  float* x2    = f;
  float* rel   = x2 + SZ;
  float* qb_   = rel + NN * HEADS;
  float* kb_   = qb_ + SZ;
  float* vb_   = kb_ + SZ;
  float* attn  = vb_ + SZ;
  float* proj  = attn + SZ;
  float* xa    = proj + SZ;
  float* h1    = xa + SZ;
  float* h2    = h1 + SZ;
  float* xnext = h2 + SZ;

  auto gemm = [&](const float* X_, const float* W_, const float* bias, float* Y_, int act) {
    gemm64_kernel<<<dim3(M / 128), 256, 0, stream>>>(X_, W_, bias, Y_, M, act);
  };

  // stage 1: graph feature extraction
  topk_kernel<<<dim3(N), 256, 0, stream>>>(crn_lon, crn_lat, idxp, N);
  gather_kernel<<<dim3(M), 64, 0, stream>>>(x, cab_lon, idxp, conv_w, ln_in_g, ln_in_b, x2, N);
  rpe_kernel<<<dim3((unsigned)((NN + 255) / 256)), 256, 0, stream>>>(
      crg_lon, crg_lat, rpe_w1, rpe_b1, rpe_w2, rpe_b2, rel, N);

  // block 0 (diff=True, has_ln2=True)
  gemm(x2, blk0.qw, blk0.qb, qb_, 0);
  gemm(x2, blk0.kw, blk0.kb, kb_, 0);
  gemm(x2, blk0.vw, blk0.vb, vb_, 0);
  attn_kernel<<<dim3(N / 64, HEADS, B), 128, 0, stream>>>(qb_, kb_, vb_, rel, blk0.ls, attn, N);
  gemm(attn, blk0.ow, blk0.ob, proj, 0);
  ln_kernel<<<dim3(M), 64, 0, stream>>>(x2, proj, blk0.ln1_g, blk0.ln1_b, xa);
  gemm(xa, blk0.w1, blk0.b1, h1, 1);
  gemm(h1, blk0.w2, blk0.b2, h2, 1);
  ln_kernel<<<dim3(M), 64, 0, stream>>>(xa, h2, blk0.ln2_g, blk0.ln2_b, xnext);

  // block 1 (diff=False, has_ln2=False, out_dim=1)
  gemm(xnext, blk1.qw, blk1.qb, qb_, 0);
  gemm(xnext, blk1.kw, blk1.kb, kb_, 0);
  gemm(xnext, blk1.vw, blk1.vb, vb_, 0);
  attn_kernel<<<dim3(N / 64, HEADS, B), 128, 0, stream>>>(qb_, kb_, vb_, rel, blk1.ls, attn, N);
  gemm(attn, blk1.ow, blk1.ob, proj, 0);
  ln_kernel<<<dim3(M), 64, 0, stream>>>(xnext, proj, blk1.ln1_g, blk1.ln1_b, xa);
  gemm(xa, blk1.w1, blk1.b1, h1, 1);
  gemv_relu_kernel<<<dim3(M / 8), 256, 0, stream>>>(h1, blk1.w2, blk1.b2, (float*)d_out);
}
